// ResNet_ForCheck_72284299592421
// MI455X (gfx1250) — matrix operand order repaired and checked by reading, not compiled, not run on hardware
//
#include <hip/hip_runtime.h>
#include <hip/hip_bf16.h>

typedef _Float16 h16 __attribute__((ext_vector_type(16)));
typedef _Float16 h8  __attribute__((ext_vector_type(8)));
typedef float    v8f __attribute__((ext_vector_type(8)));
typedef unsigned int u32x4 __attribute__((ext_vector_type(4)));
typedef int          i32x4 __attribute__((ext_vector_type(4)));
typedef int          i32x8 __attribute__((ext_vector_type(8)));

#define BN_EPS 1e-5f

// ---------------------------------------------------------------------------
// Repack OIHW f32 weights into WMMA B-fragment order, f16.
// dst index t = (((ntile*nK + ks)*32 + lane)*16 + e)
//   k  = ks*32 + (e>>3)*16 + (lane>>4)*8 + (e&7)   (GEMM K index, K = KH*KW*C, C-contig)
//   oc = ntile*16 + (lane&15)
// ---------------------------------------------------------------------------
__global__ void repack_weights(const float* __restrict__ w, _Float16* __restrict__ dst,
                               int OC, int C, int KH, int KW) {
    long K = (long)KH * KW * C;
    long total = (long)OC * K;
    long t = (long)blockIdx.x * blockDim.x + threadIdx.x;
    if (t >= total) return;
    int e    = (int)(t & 15);
    long r   = t >> 4;
    int lane = (int)(r & 31);
    r >>= 5;
    int nK   = (int)(K / 32);
    int ks   = (int)(r % nK);
    int ntile= (int)(r / nK);
    int k    = ks * 32 + ((e >> 3) * 16) + ((lane >> 4) * 8) + (e & 7);
    int oc   = ntile * 16 + (lane & 15);
    int c    = k % C;
    int patch= k / C;
    int kh   = patch / KW;
    int kw   = patch % KW;
    dst[t] = (_Float16)w[(((long)oc * C + c) * KH + kh) * KW + kw];
}

// ---------------------------------------------------------------------------
// Fold eval-mode BN into scale/bias: y = x*scale + bias
// ---------------------------------------------------------------------------
__global__ void bn_fuse(const float* __restrict__ g, const float* __restrict__ b,
                        const float* __restrict__ m, const float* __restrict__ v,
                        float* __restrict__ scale, float* __restrict__ bias, int C) {
    int c = blockIdx.x * blockDim.x + threadIdx.x;
    if (c < C) {
        float s = g[c] * rsqrtf(v[c] + BN_EPS);
        scale[c] = s;
        bias[c]  = b[c] - m[c] * s;
    }
}

// ---------------------------------------------------------------------------
// conv1: 3x3, C=3, stride 1, pad 1, N=256, H=W=32, OC=64.
// f32 NCHW input -> f16 padded-NHWC (34x34 halo) output, BN + ReLU fused.
// ---------------------------------------------------------------------------
__global__ void conv1_direct(const float* __restrict__ x, const float* __restrict__ w,
                             const float* __restrict__ scale, const float* __restrict__ bias,
                             _Float16* __restrict__ out) {
    long t = (long)blockIdx.x * blockDim.x + threadIdx.x; // = ((n*32+oh)*32+ow)*64+oc
    int oc = (int)(t & 63);
    long r = t >> 6;
    int ow = (int)(r & 31); r >>= 5;
    int oh = (int)(r & 31);
    int n  = (int)(r >> 5);
    float acc = 0.f;
    #pragma unroll
    for (int c = 0; c < 3; ++c)
        #pragma unroll
        for (int kh = 0; kh < 3; ++kh) {
            int ih = oh + kh - 1;
            if (ih < 0 || ih >= 32) continue;
            #pragma unroll
            for (int kw = 0; kw < 3; ++kw) {
                int iw = ow + kw - 1;
                if (iw < 0 || iw >= 32) continue;
                acc += x[(((long)n * 3 + c) * 32 + ih) * 32 + iw] *
                       w[((oc * 3 + c) * 3 + kh) * 3 + kw];
            }
        }
    float vv = acc * scale[oc] + bias[oc];
    long oidx = (((long)n * 34 + oh + 1) * 34 + ow + 1) * 64 + oc;
    out[oidx] = (_Float16)fmaxf(vv, 0.f);
}

__device__ __forceinline__ h16 loadA(const _Float16* p) {
    h8 lo = *(const h8*)p;          // K = base + 0..7
    h8 hf = *(const h8*)(p + 16);   // K = base + 16..23
    return __builtin_shufflevector(lo, hf, 0,1,2,3,4,5,6,7,8,9,10,11,12,13,14,15);
}

// ---------------------------------------------------------------------------
// Tensor Data Mover: DMA a contiguous panel (numUnits x 8 bytes) from global
// memory into LDS at byte offset ldsByteOff. D# packing per CDNA5 ISA
// CDNA5 ISA (group0: count/lds_addr/global_addr/type=2; group1:
// data_size=8B, 1-D tile: tensor_dim0=tile_dim0=numUnits, dims1=1).
// Issued by one wave per workgroup; completion via s_wait_tensorcnt.
// ---------------------------------------------------------------------------
__device__ __forceinline__ void tdm_load_to_lds(const void* gaddr, unsigned ldsByteOff,
                                                unsigned numUnits) {
    unsigned long long ga = (unsigned long long)gaddr;
    u32x4 g0 = {};
    g0[0] = 1u;                                          // count=1 (valid), user mode
    g0[1] = ldsByteOff;                                  // lds_addr [63:32]
    g0[2] = (unsigned)(ga & 0xffffffffu);                // global_addr [95:64]
    g0[3] = (unsigned)((ga >> 32) & 0x01ffffffu)         // global_addr [120:96]
          | (2u << 30);                                  // type=2 ("image") [127:126]
    i32x8 g1 = {};
    g1[0] = (3 << 16);                                   // data_size=3 (8B), wg_mask=0
    g1[1] = (int)((numUnits & 0xffffu) << 16);           // tensor_dim0[15:0]  @ [63:48]
    g1[2] = (int)((numUnits >> 16) & 0xffffu)            // tensor_dim0[31:16] @ [79:64]
          | (1 << 16);                                   // tensor_dim1=1      @ [111:80]
    g1[3] = (int)((numUnits & 0xffffu) << 16);           // tile_dim0          @ [127:112]
    g1[4] = 1;                                           // tile_dim1=1        @ [143:128]
    g1[5] = (int)numUnits;                               // tensor_dim0_stride @ [207:160]
    g1[6] = 0;
    g1[7] = 0;
    i32x4 z4 = {};
#if defined(__clang_major__) && (__clang_major__ >= 23)
    i32x8 z8 = {};
    __builtin_amdgcn_tensor_load_to_lds(g0, g1, z4, z4, z8, 0);
#else
    __builtin_amdgcn_tensor_load_to_lds(g0, g1, z4, z4, 0);
#endif
}

// ---------------------------------------------------------------------------
// Implicit-GEMM conv via WMMA f32_16x16x32_f16, zero-padded NHWC activations
// (spatial halo of 1 -> no boundary predication anywhere in the hot loop).
// One wave computes a 32x16 (M x OC) tile: two accumulators share each B
// fragment. 8 waves/block stacked along M (256 rows/block).
// The ENTIRE B panel for this OC tile (nK * 1KB, <= 147 KB) is DMA'd once
// per block into dynamic LDS by the Tensor Data Mover; the K loop then runs
// barrier-free with ds_load_b128 B reads.
// Manual software pipeline: chunk i+1 loads issue before chunk i WMMAs.
// Epilogue: BN scale/bias, optional residual add, optional ReLU, padded out.
// ---------------------------------------------------------------------------
__global__ __launch_bounds__(256) void wmma_conv(
    const _Float16* __restrict__ in, const _Float16* __restrict__ wk,
    const float* __restrict__ scale, const float* __restrict__ bias,
    const _Float16* __restrict__ res, _Float16* __restrict__ out,
    int H, int W, int C, int OC, int OH, int OW,
    int KH, int KW, int strd, int pad, int relu, int hasres) {

    extern __shared__ _Float16 bsh[];        // nK*512 halves, dynamic (<=147 KB)

    const int tid  = threadIdx.x;
    const int lane = tid & 31;
    const int wv   = tid >> 5;
    const long tile_m = ((long)blockIdx.x * 8 + wv) * 32;
    const int  tile_n = blockIdx.y * 16;
    const int  row = lane & 15;
    const int  hi  = lane >> 4;
    const int  nKc = C >> 5;                 // 32-wide K chunks per (kh,kw)
    const int  nK  = KH * KW * nKc;
    const int  HP = H + 2, WP = W + 2, OHP = OH + 2, OWP = OW + 2;
    const int  ohw = OH * OW;

    const _Float16* wbase = wk + (long)(tile_n >> 4) * nK * 512;  // 512 halves/kstep

    // ---- one-shot TDM stage of the whole B panel into LDS ----
    if (wv == 0) {
        tdm_load_to_lds(wbase, 0u, (unsigned)(nK * 128));  // nK * 1KB in 8B units
        __builtin_amdgcn_s_wait_tensorcnt(0);
    }
    __syncthreads();

    // Two A rows per lane (fragment 0: rows 0..15, fragment 1: rows 16..31).
    const long m0 = tile_m + row;
    const long m1 = m0 + 16;
    const int n0 = (int)(m0 / ohw), r0 = (int)(m0 % ohw), oh0 = r0 / OW, ow0 = r0 % OW;
    const int n1 = (int)(m1 / ohw), r1 = (int)(m1 % ohw), oh1 = r1 / OW, ow1 = r1 % OW;
    const _Float16* abase0 =
        in + (((long)n0 * HP + (oh0 * strd - pad + 1)) * WP + (ow0 * strd - pad + 1)) * C + hi * 8;
    const _Float16* abase1 =
        in + (((long)n1 * HP + (oh1 * strd - pad + 1)) * WP + (ow1 * strd - pad + 1)) * C + hi * 8;

    v8f acc0 = {};
    v8f acc1 = {};
    int ks = 0;
    for (int kh = 0; kh < KH; ++kh) {
        // prefetch next input row for this lane's two fragments
        if (kh + 1 < KH) {
            __builtin_prefetch((const char*)(abase0 + ((long)(kh + 1) * WP) * C), 0, 0);
            __builtin_prefetch((const char*)(abase1 + ((long)(kh + 1) * WP) * C), 0, 0);
        }
        for (int kw = 0; kw < KW; ++kw, ks += nKc) {
            const _Float16* p0 = abase0 + ((long)kh * WP + kw) * C;
            const _Float16* p1 = abase1 + ((long)kh * WP + kw) * C;
            const _Float16* bp = bsh + (long)ks * 512 + lane * 16;

            // ---- software-pipelined K loop: unconditional b128 loads ----
            h16 a0 = loadA(p0);
            h16 a1 = loadA(p1);
            h16 b  = *(const h16*)bp;
            for (int cb = 0; cb + 1 < nKc; ++cb) {
                p0 += 32; p1 += 32; bp += 512;
                h16 a0n = loadA(p0);
                h16 a1n = loadA(p1);
                h16 bn  = *(const h16*)bp;
                acc0 = __builtin_amdgcn_wmma_f32_16x16x32_f16(
                           false, a0, false, b, (short)0, acc0, false, false);
                acc1 = __builtin_amdgcn_wmma_f32_16x16x32_f16(
                           false, a1, false, b, (short)0, acc1, false, false);
                a0 = a0n; a1 = a1n; b = bn;
            }
            acc0 = __builtin_amdgcn_wmma_f32_16x16x32_f16(
                       false, a0, false, b, (short)0, acc0, false, false);
            acc1 = __builtin_amdgcn_wmma_f32_16x16x32_f16(
                       false, a1, false, b, (short)0, acc1, false, false);
        }
    }

    // D layout: VGPR j -> M = j + hi*8 (+16 for frag 1), N = lane&15.
    const int   occ = tile_n + row;
    const float s   = scale[occ];
    const float bb  = bias[occ];
    #pragma unroll
    for (int f = 0; f < 2; ++f) {
        const v8f a = f ? acc1 : acc0;
        long mmb = tile_m + f * 16 + hi * 8;
        int n2  = (int)(mmb / ohw);
        int rr  = (int)(mmb % ohw);
        int oh2 = rr / OW;
        int ow2 = rr % OW;
        #pragma unroll
        for (int j = 0; j < 8; ++j) {
            const long oidx = (((long)n2 * OHP + oh2 + 1) * OWP + ow2 + 1) * OC + occ;
            float vv = a[j] * s + bb;
            if (hasres) vv += (float)res[oidx];
            if (relu)   vv = fmaxf(vv, 0.f);
            out[oidx] = (_Float16)vv;
            if (++ow2 == OW) { ow2 = 0; if (++oh2 == OH) { oh2 = 0; ++n2; } }
        }
    }
}

// ---------------------------------------------------------------------------
// Stats pass 1: coalesced h8 reads over the whole (padded) buffer. Thread
// stride is 131072 elements (64 blocks * 256 thr * 8), a multiple of every C,
// so each thread privately accumulates a fixed set of 8 channels -> no
// atomics, bitwise deterministic. Pad elements are zero and don't perturb
// the sums. part[b*2048 + t*8 + e] corresponds to channel (b*2048+t*8+e)%C.
// Requires only E % 8 == 0.
// ---------------------------------------------------------------------------
__global__ __launch_bounds__(256) void channel_sum_p1(const _Float16* __restrict__ act,
                                                      float* __restrict__ part, long E) {
    const int  b  = blockIdx.x;              // 64 blocks
    const int  t  = threadIdx.x;             // 256 threads
    const long g0 = ((long)b * 256 + t) * 8;
    float s[8] = {0.f, 0.f, 0.f, 0.f, 0.f, 0.f, 0.f, 0.f};
    for (long off = g0; off + 8 <= E; off += 131072) {
        h8 x = *(const h8*)(act + off);
        #pragma unroll
        for (int e = 0; e < 8; ++e) s[e] += (float)x[e];
    }
    float* dst = part + b * 2048 + t * 8;
    #pragma unroll
    for (int e = 0; e < 8; ++e) dst[e] = s[e];
}

// ---------------------------------------------------------------------------
// Stats pass 2: deterministic reduction of 64*(2048/C) partials per channel.
// grid = C blocks, 64 threads.
// ---------------------------------------------------------------------------
__global__ void channel_sum_p2(const float* __restrict__ part, float* __restrict__ out,
                               int C) {
    const int c = blockIdx.x;
    const int t = threadIdx.x;               // 64 threads (one per pass-1 block)
    const int reps = 2048 / C;
    float s = 0.f;
    for (int j = 0; j < reps; ++j)
        s += part[t * 2048 + j * C + c];
    __shared__ float sm[64];
    sm[t] = s;
    __syncthreads();
    for (int off = 32; off > 0; off >>= 1) {
        if (t < off) sm[t] += sm[t + off];
        __syncthreads();
    }
    if (t == 0) out[c] = sm[0];
}

// ---------------------------------------------------------------------------
// Global avg-pool (4x4) + linear head, padded 6x6x512 input. Block per batch.
// ---------------------------------------------------------------------------
__global__ void head_kernel(const _Float16* __restrict__ act, const float* __restrict__ lw,
                            const float* __restrict__ lb, float* __restrict__ logits) {
    const int nb = blockIdx.x;
    __shared__ float pooled[512];
    for (int c = threadIdx.x; c < 512; c += blockDim.x) {
        float s = 0.f;
        #pragma unroll
        for (int y = 0; y < 4; ++y)
            #pragma unroll
            for (int x = 0; x < 4; ++x)
                s += (float)act[(((long)nb * 6 + y + 1) * 6 + x + 1) * 512 + c];
        pooled[c] = s * (1.f / 16.f);
    }
    __syncthreads();
    if ((int)threadIdx.x < 10) {
        float s = lb[threadIdx.x];
        for (int c = 0; c < 512; ++c)
            s += pooled[c] * lw[threadIdx.x * 512 + c];
        logits[nb * 10 + threadIdx.x] = s;
    }
}

// ---------------------------------------------------------------------------
// Host orchestration
// ---------------------------------------------------------------------------
extern "C" void kernel_launch(void* const* d_in, const int* in_sizes, int n_in,
                              void* d_out, int out_size, void* d_ws, size_t ws_size,
                              hipStream_t stream) {
    (void)in_sizes; (void)n_in; (void)out_size; (void)ws_size;

    const float* X = (const float*)d_in[0];
    auto F = [&](int i) { return (const float*)d_in[i]; };

    char* ws = (char*)d_ws;
    const size_t ACT = 40ull << 20;                 // 40 MB per activation buffer
    _Float16* bufA  = (_Float16*)(ws);
    _Float16* bufB  = (_Float16*)(ws + ACT);
    _Float16* bufC  = (_Float16*)(ws + 2 * ACT);
    _Float16* wpack = (_Float16*)(ws + 3 * ACT);            // ~11.2M halves (~22.3 MB) needed
    float*    bnbuf = (float*)(ws + 3 * ACT + (24ull << 20));
    float*    part  = (float*)(ws + 3 * ACT + (25ull << 20)); // 64*2048 floats
    float*    out   = (float*)d_out;

    size_t wOff = 0, bnOff = 0;
    long geomB[3] = {0, 0, 0};   // bytes of current padded geometry per buffer slot

    // Zero a buffer's pad ring (whole buffer) only when its geometry changes.
    auto ensure = [&](void* p, long bytes) {
        int slot = (int)(((char*)p - ws) / (long)ACT);
        if (slot >= 0 && slot < 3 && geomB[slot] != bytes) {
            hipMemsetAsync(p, 0, (size_t)bytes, stream);
            geomB[slot] = bytes;
        }
    };

    auto conv = [&](int widx, int bnidx, const _Float16* in, const _Float16* rsd,
                    _Float16* dst, int H, int W, int C, int OC,
                    int KH, int KW, int strd, int pad, int relu) {
        const long K = (long)KH * KW * C;
        const long total = (long)OC * K;
        _Float16* wp = wpack + wOff; wOff += (size_t)total;
        float*    sb = bnbuf + bnOff; bnOff += (size_t)2 * OC;
        repack_weights<<<dim3((unsigned)((total + 255) / 256)), 256, 0, stream>>>(
            F(widx), wp, OC, C, KH, KW);
        bn_fuse<<<dim3((unsigned)((OC + 63) / 64)), 64, 0, stream>>>(
            F(bnidx), F(bnidx + 1), F(bnidx + 2), F(bnidx + 3), sb, sb + OC, OC);
        const int OH = (H + 2 * pad - KH) / strd + 1;
        const int OW = (W + 2 * pad - KW) / strd + 1;
        const long M = 256L * OH * OW;              // always a multiple of 256 here
        ensure(dst, 256L * (OH + 2) * (OW + 2) * OC * 2);
        dim3 grid((unsigned)(M / 256), (unsigned)(OC / 16));
        const size_t shmem = (size_t)(K / 32) * 1024;   // whole B panel, <=147 KB
        wmma_conv<<<grid, 256, shmem, stream>>>(in, wp, sb, sb + OC, rsd, dst,
                                                H, W, C, OC, OH, OW,
                                                KH, KW, strd, pad, relu, rsd ? 1 : 0);
    };
    auto stats = [&](const _Float16* act, float* dst, int C, long Epad) {
        channel_sum_p1<<<64, 256, 0, stream>>>(act, part, Epad);
        channel_sum_p2<<<dim3((unsigned)C), 64, 0, stream>>>(part, dst, C);
    };
    auto swapAC = [&]() { _Float16* t = bufA; bufA = bufC; bufC = t; };

    // conv1 + bn1 + relu  (x f32 NCHW -> bufA padded f16 NHWC 34x34x64)
    {
        float* sb = bnbuf + bnOff; bnOff += 128;
        bn_fuse<<<1, 64, 0, stream>>>(F(2), F(3), F(4), F(5), sb, sb + 64, 64);
        ensure(bufA, 256L * 34 * 34 * 64 * 2);
        const long total = 256L * 32 * 32 * 64;
        conv1_direct<<<dim3((unsigned)(total / 256)), 256, 0, stream>>>(
            X, F(1), sb, sb + 64, bufA);
    }
    const long E0 = 256L * 34 * 34 * 64;
    const long E1 = 256L * 18 * 18 * 128;
    const long E2 = 256L * 10 * 10 * 256;
    const long E3 = 256L * 6 * 6 * 512;
    stats(bufA, out + 2560, 64, E0);

    // ---- stage 0: 64ch, 32x32, stride 1, identity shortcuts ----
    conv( 6,  7, bufA, nullptr, bufB, 32, 32, 64,  64, 3, 3, 1, 1, 1);
    conv(11, 12, bufB, bufA,    bufC, 32, 32, 64,  64, 3, 3, 1, 1, 1);
    swapAC();
    conv(16, 17, bufA, nullptr, bufB, 32, 32, 64,  64, 3, 3, 1, 1, 1);
    conv(21, 22, bufB, bufA,    bufC, 32, 32, 64,  64, 3, 3, 1, 1, 1);
    swapAC();
    stats(bufA, out + 2624, 64, E0);

    // ---- stage 1: 64 -> 128, 32x32 -> 16x16, projection shortcut ----
    conv(26, 27, bufA, nullptr, bufB, 32, 32, 64, 128, 3, 3, 2, 1, 1);
    conv(36, 37, bufA, nullptr, bufC, 32, 32, 64, 128, 1, 1, 2, 0, 0);
    conv(31, 32, bufB, bufC,    bufA, 16, 16, 128, 128, 3, 3, 1, 1, 1);
    conv(41, 42, bufA, nullptr, bufB, 16, 16, 128, 128, 3, 3, 1, 1, 1);
    conv(46, 47, bufB, bufA,    bufC, 16, 16, 128, 128, 3, 3, 1, 1, 1);
    swapAC();
    stats(bufA, out + 2688, 128, E1);

    // ---- stage 2: 128 -> 256, 16x16 -> 8x8 ----
    conv(51, 52, bufA, nullptr, bufB, 16, 16, 128, 256, 3, 3, 2, 1, 1);
    conv(61, 62, bufA, nullptr, bufC, 16, 16, 128, 256, 1, 1, 2, 0, 0);
    conv(56, 57, bufB, bufC,    bufA, 8, 8, 256, 256, 3, 3, 1, 1, 1);
    conv(66, 67, bufA, nullptr, bufB, 8, 8, 256, 256, 3, 3, 1, 1, 1);
    conv(71, 72, bufB, bufA,    bufC, 8, 8, 256, 256, 3, 3, 1, 1, 1);
    swapAC();
    stats(bufA, out + 2816, 256, E2);

    // ---- stage 3: 256 -> 512, 8x8 -> 4x4 ----
    conv(76, 77, bufA, nullptr, bufB, 8, 8, 256, 512, 3, 3, 2, 1, 1);
    conv(86, 87, bufA, nullptr, bufC, 8, 8, 256, 512, 1, 1, 2, 0, 0);
    conv(81, 82, bufB, bufC,    bufA, 4, 4, 512, 512, 3, 3, 1, 1, 1);
    conv(91, 92, bufA, nullptr, bufB, 4, 4, 512, 512, 3, 3, 1, 1, 1);
    conv(96, 97, bufB, bufA,    bufC, 4, 4, 512, 512, 3, 3, 1, 1, 1);
    swapAC();
    stats(bufA, out + 3072, 512, E3);

    // ---- avg-pool + linear head ----
    head_kernel<<<256, 256, 0, stream>>>(bufA, F(101), F(102), out);
}
